// HeteroGNN_51041391345810
// MI455X (gfx1250) — matrix operand order repaired and checked by reading, not compiled, not run on hardware
//
#include <hip/hip_runtime.h>
#include <hip/hip_bf16.h>
#include <math.h>

// ---------------------------------------------------------------------------
// HeteroGNN (GATv2) forward for MI455X (gfx1250, wave32).
// GEMMs use v_wmma_f32_16x16x32_bf16 (f32 accumulate). Edge phase is
// L2-resident gather/scatter with f32 atomics (one wave per edge).
// ---------------------------------------------------------------------------

#define NEG_SLOPE 0.2f

typedef __attribute__((ext_vector_type(16))) __bf16 v16bf;
typedef __attribute__((ext_vector_type(8)))  float  v8f;

// ---------------- GEMM: C[M,64] = A[M,K] @ W[K,64] (+ bias) ----------------
// One wave per 16x16 output tile. 8 waves/block cover 32 rows x 64 cols.
// M must be a multiple of 32 (true here: 200000 / 100000).
__global__ void __launch_bounds__(256)
k_gemm_bf16(const float* __restrict__ A, const float* __restrict__ W,
            const float* __restrict__ bias, float* __restrict__ C,
            int M, int K)
{
    const int lane  = threadIdx.x & 31;
    const int wave  = threadIdx.x >> 5;
    const int ntile = wave & 3;                     // 4 col tiles of 16
    const int m0    = (blockIdx.x * 2 + (wave >> 2)) * 16;
    if (m0 >= M) return;                            // wave-uniform

    const int row  = m0 + (lane & 15);              // A row for this lane
    const int aoff = (lane >> 4) << 3;              // 0 or 8  (ISA A 16x32 layout)
    const int col  = ntile * 16 + (lane & 15);      // B/C column for this lane
    const int boff = (lane >> 4) << 3;              // 0 or 8  (same K order as A)

    v8f acc = {};
    const float* __restrict__ ap = A + (size_t)row * K;
    for (int k0 = 0; k0 < K; k0 += 32) {
        v16bf a, b;
        // A fragment: elem e<8 -> K = k0+aoff+e ; e>=8 -> K = k0+aoff+e+8
#pragma unroll
        for (int e = 0; e < 8; ++e)  a[e] = (__bf16)ap[k0 + aoff + e];
#pragma unroll
        for (int e = 8; e < 16; ++e) a[e] = (__bf16)ap[k0 + aoff + e + 8];
        // B fragment: lane holds one column, elem e<8 -> K = k0+boff+e ; e>=8 -> K = k0+boff+e+8
#pragma unroll
        for (int e = 0; e < 16; ++e)
            b[e] = (__bf16)W[(size_t)(k0 + boff + e + (e & 8)) * 64 + col];
        acc = __builtin_amdgcn_wmma_f32_16x16x32_bf16(
                  false, a, false, b, (short)0, acc, false, false);
    }
    const float bv   = bias ? bias[col] : 0.0f;
    const int   crow = m0 + ((lane >> 4) << 3);     // C/D: vgpr r -> row crow+r
#pragma unroll
    for (int r = 0; r < 8; ++r)
        C[(size_t)(crow + r) * 64 + col] = acc[r] + bv;
}

// ---------------- fill --------------------------------------------------
__global__ void k_fill(float* __restrict__ p, float v, int n)
{
    int i = blockIdx.x * blockDim.x + threadIdx.x;
    if (i < n) p[i] = v;
}

// new[node,c] = b0[c] + b1[c]   (HeteroConv 'sum' bias pre-accumulation)
__global__ void k_init_bias2(float* __restrict__ out, const float* __restrict__ b0,
                             const float* __restrict__ b1, int N)
{
    int i = blockIdx.x * blockDim.x + threadIdx.x;
    if (i >= N * 64) return;
    int c = i & 63;
    out[i] = b0[c] + (b1 ? b1[c] : 0.0f);
}

// ---------------- edge phase --------------------------------------------
// One wave per edge; lane = channel. logits[e,h] = sum_c lrelu(xl+xr)*att.
__global__ void __launch_bounds__(256)
k_edge_logits(const float* __restrict__ xl, const float* __restrict__ xr,
              const int* __restrict__ src, const int* __restrict__ dst,
              const float* __restrict__ att, float* __restrict__ logits,
              float* __restrict__ mmax, int E)
{
    int gid  = blockIdx.x * blockDim.x + threadIdx.x;
    int e    = gid >> 5;
    int lane = gid & 31;
    if (e >= E) return;
    int s = src[e], d = dst[e];
    const float* pl = xl + (size_t)s * 64;
    const float* pr = xr + (size_t)d * 64;
    float a0 = pl[lane]      + pr[lane];
    float a1 = pl[lane + 32] + pr[lane + 32];
    a0 = (a0 > 0.0f) ? a0 : a0 * NEG_SLOPE;
    a1 = (a1 > 0.0f) ? a1 : a1 * NEG_SLOPE;
    float t0 = a0 * att[lane];
    float t1 = a1 * att[lane + 32];
#pragma unroll
    for (int off = 16; off; off >>= 1) {
        t0 += __shfl_xor(t0, off, 32);
        t1 += __shfl_xor(t1, off, 32);
    }
    if (lane == 0) {
        logits[(size_t)e * 2]     = t0;
        logits[(size_t)e * 2 + 1] = t1;
        atomicMax(&mmax[d * 2],     t0);
        atomicMax(&mmax[d * 2 + 1], t1);
    }
}

// one thread per (edge, head): ex = exp(logit - m[dst]); den[dst] += ex
__global__ void k_edge_exp(const int* __restrict__ dst, float* __restrict__ logits,
                           const float* __restrict__ mmax, float* __restrict__ den,
                           int E)
{
    int gid = blockIdx.x * blockDim.x + threadIdx.x;
    if (gid >= 2 * E) return;
    int e = gid >> 1, h = gid & 1;
    int d = dst[e];
    float v = __expf(logits[gid] - mmax[d * 2 + h]);
    logits[gid] = v;
    atomicAdd(&den[d * 2 + h], v);
}

// one wave per edge: out[dst] += xl[src] * alpha
__global__ void __launch_bounds__(256)
k_edge_agg(const float* __restrict__ xl, const int* __restrict__ src,
           const int* __restrict__ dst, const float* __restrict__ ex,
           const float* __restrict__ den, float* __restrict__ out, int E)
{
    int gid  = blockIdx.x * blockDim.x + threadIdx.x;
    int e    = gid >> 5;
    int lane = gid & 31;
    if (e >= E) return;
    int s = src[e], d = dst[e];
    float a0 = ex[(size_t)e * 2]     / den[d * 2];
    float a1 = ex[(size_t)e * 2 + 1] / den[d * 2 + 1];
    const float* pl = xl + (size_t)s * 64;
    atomicAdd(&out[(size_t)d * 64 + lane],      pl[lane]      * a0);
    atomicAdd(&out[(size_t)d * 64 + 32 + lane], pl[lane + 32] * a1);
}

// ---------------- LayerNorm(64) + ReLU, one wave per node ----------------
__global__ void __launch_bounds__(256)
k_ln_relu(const float* __restrict__ in, const float* __restrict__ g,
          const float* __restrict__ b, float* __restrict__ out, int N)
{
    int gid  = blockIdx.x * blockDim.x + threadIdx.x;
    int node = gid >> 5;
    int lane = gid & 31;
    if (node >= N) return;
    float v0 = in[(size_t)node * 64 + lane];
    float v1 = in[(size_t)node * 64 + 32 + lane];
    float s = v0 + v1;
#pragma unroll
    for (int off = 16; off; off >>= 1) s += __shfl_xor(s, off, 32);
    float mu = s * (1.0f / 64.0f);
    float d0 = v0 - mu, d1 = v1 - mu;
    float q = d0 * d0 + d1 * d1;
#pragma unroll
    for (int off = 16; off; off >>= 1) q += __shfl_xor(q, off, 32);
    float inv = rsqrtf(q * (1.0f / 64.0f) + 1e-5f);
    float o0 = d0 * inv * g[lane]      + b[lane];
    float o1 = d1 * inv * g[lane + 32] + b[lane + 32];
    out[(size_t)node * 64 + lane]      = fmaxf(o0, 0.0f);
    out[(size_t)node * 64 + 32 + lane] = fmaxf(o1, 0.0f);
}

// ---------------------------------------------------------------------------
extern "C" void kernel_launch(void* const* d_in, const int* in_sizes, int n_in,
                              void* d_out, int out_size, void* d_ws, size_t ws_size,
                              hipStream_t stream)
{
    const int NP = in_sizes[0] / 128;   // 200000
    const int NA = in_sizes[1] / 128;   // 100000
    const int E  = in_sizes[2];         // 1000000

    const float* x_p = (const float*)d_in[0];
    const float* x_a = (const float*)d_in[1];
    const int* e_src[3] = {(const int*)d_in[2], (const int*)d_in[4], (const int*)d_in[6]};
    const int* e_dst[3] = {(const int*)d_in[3], (const int*)d_in[5], (const int*)d_in[7]};
    const float* inW_p = (const float*)d_in[8];
    const float* inW_a = (const float*)d_in[9];
    const float* Wl    = (const float*)d_in[10];
    const float* bl    = (const float*)d_in[11];
    const float* Wr    = (const float*)d_in[12];
    const float* br    = (const float*)d_in[13];
    const float* att   = (const float*)d_in[14];
    const float* gbias = (const float*)d_in[15];
    const float* lng   = (const float*)d_in[16];
    const float* lnb   = (const float*)d_in[17];
    const float* oWp   = (const float*)d_in[18];
    const float* obp   = (const float*)d_in[19];
    const float* oWa   = (const float*)d_in[20];
    const float* oba   = (const float*)d_in[21];
    float* out = (float*)d_out;

    // workspace carve (f32): ~267 MB total
    float* ws = (float*)d_ws;
    size_t o = 0;
    float* h_p = ws + o; o += (size_t)NP * 64;
    float* h_a = ws + o; o += (size_t)NA * 64;
    float* n_p = ws + o; o += (size_t)NP * 64;
    float* n_a = ws + o; o += (size_t)NA * 64;
    float* xl  = ws + o; o += (size_t)NP * 64;
    float* xr  = ws + o; o += (size_t)NP * 64;
    float* elg = ws + o; o += (size_t)E  * 2;
    float* mx  = ws + o; o += (size_t)NP * 2;
    float* den = ws + o; o += (size_t)NP * 2;
    (void)ws_size; (void)n_in; (void)out_size;

    const dim3 blk(256);
    const float NEG_INF = -__builtin_huge_valf();

    // input projections (K = 128)
    k_gemm_bf16<<<NP / 32, blk, 0, stream>>>(x_p, inW_p, nullptr, h_p, NP, 128);
    k_gemm_bf16<<<NA / 32, blk, 0, stream>>>(x_a, inW_a, nullptr, h_a, NA, 128);

    for (int l = 0; l < 2; ++l) {
        // new[dt] pre-loaded with summed per-relation biases
        k_init_bias2<<<(NP * 64 + 255) / 256, blk, 0, stream>>>(
            n_p, gbias + (l * 3 + 0) * 64, gbias + (l * 3 + 2) * 64, NP);
        k_init_bias2<<<(NA * 64 + 255) / 256, blk, 0, stream>>>(
            n_a, gbias + (l * 3 + 1) * 64, nullptr, NA);

        for (int r = 0; r < 3; ++r) {
            const float *hs, *hd; float* nb; int Ns, Nd;
            if (r == 0)      { hs = h_a; Ns = NA; hd = h_p; Nd = NP; nb = n_p; } // writes: author->paper
            else if (r == 1) { hs = h_p; Ns = NP; hd = h_a; Nd = NA; nb = n_a; } // rev:    paper->author
            else             { hs = h_p; Ns = NP; hd = h_p; Nd = NP; nb = n_p; } // cites:  paper->paper
            const int rr = l * 3 + r;

            k_gemm_bf16<<<Ns / 32, blk, 0, stream>>>(hs, Wl + (size_t)rr * 4096,
                                                     bl + rr * 64, xl, Ns, 64);
            k_gemm_bf16<<<Nd / 32, blk, 0, stream>>>(hd, Wr + (size_t)rr * 4096,
                                                     br + rr * 64, xr, Nd, 64);
            k_fill<<<(Nd * 2 + 255) / 256, blk, 0, stream>>>(mx,  NEG_INF, Nd * 2);
            k_fill<<<(Nd * 2 + 255) / 256, blk, 0, stream>>>(den, 0.0f,    Nd * 2);

            k_edge_logits<<<E / 8, blk, 0, stream>>>(xl, xr, e_src[r], e_dst[r],
                                                     att + rr * 64, elg, mx, E);
            k_edge_exp<<<(2 * E + 255) / 256, blk, 0, stream>>>(e_dst[r], elg, mx, den, E);
            k_edge_agg<<<E / 8, blk, 0, stream>>>(xl, e_src[r], e_dst[r], elg, den, nb, E);
        }

        k_ln_relu<<<NP / 8, blk, 0, stream>>>(n_p, lng + (l * 2 + 0) * 64,
                                              lnb + (l * 2 + 0) * 64, h_p, NP);
        k_ln_relu<<<NA / 8, blk, 0, stream>>>(n_a, lng + (l * 2 + 1) * 64,
                                              lnb + (l * 2 + 1) * 64, h_a, NA);
    }

    // output heads -> d_out = concat(out_p, out_a)
    k_gemm_bf16<<<NP / 32, blk, 0, stream>>>(h_p, oWp, obp, out, NP, 64);
    k_gemm_bf16<<<NA / 32, blk, 0, stream>>>(h_a, oWa, oba, out + (size_t)NP * 64, NA, 64);
}
